// MultiHeadedAttention_23046794510586
// MI455X (gfx1250) — compile-verified, matrix operand order checked by reading, not run on hardware
//
#include <hip/hip_runtime.h>
#include <hip/hip_bf16.h>
#include <math.h>

#define H_  16
#define DM  1024
#define DK  64
#define B_  4
#define S_  1024

typedef __attribute__((ext_vector_type(16))) __bf16 v16bf;
typedef __attribute__((ext_vector_type(8)))  float  v8f;
typedef __attribute__((ext_vector_type(4)))  int    v4i_;
typedef __attribute__((address_space(1))) v4i_* gas_p;   // global int4*
typedef __attribute__((address_space(3))) v4i_* las_p;   // LDS int4*

union Frag { v16bf v; uint4 q[2]; unsigned short u[16]; };

__device__ __forceinline__ unsigned short f2bf(float f) {
  unsigned int u = __float_as_uint(f);
  u += 0x7fffu + ((u >> 16) & 1u);      // round-to-nearest-even to bf16
  return (unsigned short)(u >> 16);
}

// 16-byte global -> LDS copy via the gfx1250 async DMA path (bypasses VGPRs,
// tracked with ASYNCcnt); synchronous fallback if the builtin is missing.
__device__ __forceinline__ void cp16(const unsigned short* g, unsigned short* l) {
#if defined(__has_builtin) && __has_builtin(__builtin_amdgcn_global_load_async_to_lds_b128)
  __builtin_amdgcn_global_load_async_to_lds_b128(
      (gas_p)(unsigned long long)g,
      (las_p)(unsigned int)(unsigned long long)l,
      0, 0);
#else
  *(uint4*)l = *(const uint4*)g;
#endif
}

__device__ __forceinline__ void wait_async0() {
#if defined(__has_builtin) && __has_builtin(__builtin_amdgcn_s_wait_asynccnt)
  __builtin_amdgcn_s_wait_asynccnt(0);
#else
  asm volatile("s_wait_asynccnt 0" ::: "memory");
#endif
}

__global__ void cvt_bf16_kernel(const float* __restrict__ in,
                                unsigned short* __restrict__ out, int n) {
  int i = blockIdx.x * blockDim.x + threadIdx.x;
  int stride = gridDim.x * blockDim.x;
  for (; i < n; i += stride) out[i] = f2bf(in[i]);
}

// Y = A(bf16 [M=4096,K=1024]) x W(bf16 [N=1024,K=1024])^T + bias
// Block tile 128x128, BK=32, double-buffered async LDS staging.
// mode 0: fp32 out [row,col]            (final O-projection)
// mode 1: bias+RoPE -> bf16 [B,H,S,DK]  (Q and K)
// mode 2: bias -> bf16 [B,H,DK,S]       (V transposed per head)
__launch_bounds__(256)
__global__ void gemm_qkv_kernel(const unsigned short* __restrict__ A,
                                const unsigned short* __restrict__ W,
                                const float* __restrict__ bias,
                                float* __restrict__ outF,
                                unsigned short* __restrict__ outB,
                                int mode) {
  __shared__ __align__(16) unsigned short As[2][128 * 32];
  __shared__ __align__(16) unsigned short Bs[2][128 * 32];
  const int tid  = threadIdx.x;
  const int lane = tid & 31;
  const int wave = tid >> 5;
  const int wm   = wave & 1;          // 2 M-groups of 64 rows
  const int wn   = wave >> 1;         // 4 N-groups of 32 cols
  const int half = lane >> 4;
  const int l16  = lane & 15;
  const int blockN = blockIdx.x;      // 8 blocks of 128 cols
  const int blockM = blockIdx.y;      // 32 blocks of 128 rows

  const int kA0 = half ? 8 : 0;       // A/B fragment K-chunk bases (ISA layout)
  const int kA1 = half ? 24 : 16;

  // per-thread staging chunks: A and B tiles are each 512 x 16B chunks
  const int r0 = tid >> 2,          c0 = (tid & 3) << 3;
  const int r1 = (tid + 256) >> 2,  c1 = c0;
  const unsigned short* ga0 = &A[(size_t)(blockM * 128 + r0) * 1024 + c0];
  const unsigned short* ga1 = &A[(size_t)(blockM * 128 + r1) * 1024 + c1];
  const unsigned short* gb0 = &W[(size_t)(blockN * 128 + r0) * 1024 + c0];
  const unsigned short* gb1 = &W[(size_t)(blockN * 128 + r1) * 1024 + c1];
  unsigned short* la0 = &As[0][r0 * 32 + c0];
  unsigned short* la1 = &As[0][r1 * 32 + c1];
  unsigned short* lb0 = &Bs[0][r0 * 32 + c0];
  unsigned short* lb1 = &Bs[0][r1 * 32 + c1];
  const int LBUF = 128 * 32;          // elements per LDS buffer

  v8f acc[4][2] = {};

  auto compute = [&](int buf) {
    Frag a[4], b[2];
#pragma unroll
    for (int mi = 0; mi < 4; ++mi) {
      int r = wm * 64 + mi * 16 + l16;
      a[mi].q[0] = *(const uint4*)&As[buf][r * 32 + kA0];
      a[mi].q[1] = *(const uint4*)&As[buf][r * 32 + kA1];
    }
#pragma unroll
    for (int ni = 0; ni < 2; ++ni) {
      int r = wn * 32 + ni * 16 + l16;
      b[ni].q[0] = *(const uint4*)&Bs[buf][r * 32 + kA0];
      b[ni].q[1] = *(const uint4*)&Bs[buf][r * 32 + kA1];
    }
#pragma unroll
    for (int mi = 0; mi < 4; ++mi)
#pragma unroll
      for (int ni = 0; ni < 2; ++ni)
        acc[mi][ni] = __builtin_amdgcn_wmma_f32_16x16x32_bf16(
            false, a[mi].v, false, b[ni].v, (short)0, acc[mi][ni], false, false);
  };

  // prologue: stage k-step 0 into buffer 0
  cp16(ga0, la0);
  cp16(ga1, la1);
  cp16(gb0, lb0);
  cp16(gb1, lb1);

  // branchless main loop: consume buf kt&1, stage kt+1 into the other buffer
  for (int kt = 0; kt < 31; ++kt) {
    const int buf = kt & 1;
    const int nb  = buf ^ 1;
    const int ko  = (kt + 1) * 32;
    wait_async0();
    __syncthreads();                  // all waves' async stores to buf complete
    cp16(ga0 + ko, la0 + nb * LBUF);
    cp16(ga1 + ko, la1 + nb * LBUF);
    cp16(gb0 + ko, lb0 + nb * LBUF);
    cp16(gb1 + ko, lb1 + nb * LBUF);
    compute(buf);
  }
  wait_async0();
  __syncthreads();
  compute(1);                         // tail: k-step 31 lives in buffer 1

  // Epilogue.  C/D layout: element r of lane -> row r + 8*half, col l16.
#pragma unroll
  for (int mi = 0; mi < 4; ++mi) {
#pragma unroll
    for (int ni = 0; ni < 2; ++ni) {
      int col = blockN * 128 + wn * 32 + ni * 16 + l16;
      float bv = bias[col];
#pragma unroll
      for (int r = 0; r < 8; ++r) {
        int row = blockM * 128 + wm * 64 + mi * 16 + r + 8 * half;
        float t = acc[mi][ni][r] + bv;
        if (mode == 0) {
          outF[(size_t)row * 1024 + col] = t;
        } else {
          int bb = row >> 10, s = row & 1023;
          int h = col >> 6, d = col & 63;
          if (mode == 1) {
            // interleaved RoPE: pair (2i,2i+1) sits in adjacent lanes
            float other = __shfl_xor(t, 1, 32);
            float freq = __expf(-(2.0f * 6.907755278982137f / 64.0f) *
                                (float)(d >> 1));   // ln(1000)
            float sn, c;
            __sincosf((float)s * freq, &sn, &c);
            float res = (d & 1) ? (t * c + other * sn) : (t * c - other * sn);
            outB[((size_t)((bb * H_ + h) * S_ + s)) * DK + d] = f2bf(res);
          } else {
            outB[((size_t)((bb * H_ + h) * DK + d)) * S_ + s] = f2bf(t);
          }
        }
      }
    }
  }
}

// Flash attention: one block per (b,h, 64 query rows); 4 waves x 16 rows.
__launch_bounds__(128)
__global__ void flash_attn_kernel(const unsigned short* __restrict__ Q,
                                  const unsigned short* __restrict__ K,
                                  const unsigned short* __restrict__ Vt,
                                  unsigned short* __restrict__ Y) {
  __shared__ __align__(16) unsigned short kbuf[64 * 64];      // [key][dim]
  __shared__ __align__(16) unsigned short vbuf[64 * 64];      // [dim][key]
  __shared__ __align__(16) unsigned short pbuf[4 * 16 * 64];  // wave-private P
  const int tid  = threadIdx.x;
  const int lane = tid & 31;
  const int wave = tid >> 5;
  const int half = lane >> 4;
  const int l16  = lane & 15;
  const int ib   = blockIdx.x;        // query block (16)
  const int bh   = blockIdx.y;        // b*H + h (64)
  const int h    = bh & (H_ - 1);
  const int bb   = bh >> 4;
  const unsigned short* Qh = Q + (size_t)bh * S_ * DK;
  const unsigned short* Kh = K + (size_t)bh * S_ * DK;
  const unsigned short* Vh = Vt + (size_t)bh * DK * S_;

  const int qr0 = ib * 64 + wave * 16;
  const int kc0 = half ? 8 : 0;
  const int kc1 = half ? 24 : 16;

  Frag qa[2];                          // Q rows as A-fragments, d 0..31 / 32..63
  {
    size_t base = (size_t)(qr0 + l16) * DK;
    qa[0].q[0] = *(const uint4*)&Qh[base + kc0];
    qa[0].q[1] = *(const uint4*)&Qh[base + kc1];
    qa[1].q[0] = *(const uint4*)&Qh[base + 32 + kc0];
    qa[1].q[1] = *(const uint4*)&Qh[base + 32 + kc1];
  }

  float Mrow[8], Lrow[8];
  v8f o_acc[4] = {};
#pragma unroll
  for (int r = 0; r < 8; ++r) { Mrow[r] = -1e30f; Lrow[r] = 0.0f; }

  unsigned short* pw = &pbuf[wave * 16 * 64];

  for (int jb = 0; jb <= ib; ++jb) {
    __syncthreads();
#pragma unroll
    for (int i = 0; i < 4; ++i) {     // async-stage 512 x 16B chunks each
      int idx = tid + i * 128;
      int row = idx >> 3, c = (idx & 7) << 3;
      cp16(&Kh[(size_t)(jb * 64 + row) * DK + c], &kbuf[row * 64 + c]);
      cp16(&Vh[(size_t)row * S_ + jb * 64 + c], &vbuf[row * 64 + c]);
    }
    wait_async0();
    __syncthreads();

    float Sv[4][8];
#pragma unroll
    for (int n = 0; n < 4; ++n) {     // scores: S = Q K^T * 1/sqrt(dk)
      Frag bk0, bk1;
      int kr = n * 16 + l16;
      bk0.q[0] = *(const uint4*)&kbuf[kr * 64 + kc0];
      bk0.q[1] = *(const uint4*)&kbuf[kr * 64 + kc1];
      bk1.q[0] = *(const uint4*)&kbuf[kr * 64 + 32 + kc0];
      bk1.q[1] = *(const uint4*)&kbuf[kr * 64 + 32 + kc1];
      v8f sa = {};
      sa = __builtin_amdgcn_wmma_f32_16x16x32_bf16(false, qa[0].v, false, bk0.v,
                                                   (short)0, sa, false, false);
      sa = __builtin_amdgcn_wmma_f32_16x16x32_bf16(false, qa[1].v, false, bk1.v,
                                                   (short)0, sa, false, false);
#pragma unroll
      for (int r = 0; r < 8; ++r) {
        float v = sa[r] * 0.125f;
        if (jb == ib) {               // causal mask only on diagonal block
          int trow = qr0 + r + 8 * half;
          int tcol = jb * 64 + n * 16 + l16;
          if (tcol > trow) v = -1e30f;
        }
        Sv[n][r] = v;
      }
    }

    float corr[8];
#pragma unroll
    for (int r = 0; r < 8; ++r) {     // online softmax stats per query row
      float mc = fmaxf(fmaxf(Sv[0][r], Sv[1][r]), fmaxf(Sv[2][r], Sv[3][r]));
      mc = fmaxf(mc, __shfl_xor(mc, 1, 32));
      mc = fmaxf(mc, __shfl_xor(mc, 2, 32));
      mc = fmaxf(mc, __shfl_xor(mc, 4, 32));
      mc = fmaxf(mc, __shfl_xor(mc, 8, 32));
      float mn = fmaxf(Mrow[r], mc);
      corr[r] = __expf(Mrow[r] - mn);
      float ls = 0.0f;
#pragma unroll
      for (int n = 0; n < 4; ++n) {
        float p = __expf(Sv[n][r] - mn);
        Sv[n][r] = p;
        ls += p;
      }
      ls += __shfl_xor(ls, 1, 32);
      ls += __shfl_xor(ls, 2, 32);
      ls += __shfl_xor(ls, 4, 32);
      ls += __shfl_xor(ls, 8, 32);
      Lrow[r] = Lrow[r] * corr[r] + ls;
      Mrow[r] = mn;
    }
#pragma unroll
    for (int dt = 0; dt < 4; ++dt)
#pragma unroll
      for (int r = 0; r < 8; ++r) o_acc[dt][r] *= corr[r];

    // D-layout -> A-layout transpose of P via wave-private LDS
#pragma unroll
    for (int n = 0; n < 4; ++n)
#pragma unroll
      for (int r = 0; r < 8; ++r)
        pw[(r + 8 * half) * 64 + n * 16 + l16] = f2bf(Sv[n][r]);
    asm volatile("s_wait_dscnt 0" ::: "memory");

    Frag pa[2];
    pa[0].q[0] = *(const uint4*)&pw[l16 * 64 + kc0];
    pa[0].q[1] = *(const uint4*)&pw[l16 * 64 + kc1];
    pa[1].q[0] = *(const uint4*)&pw[l16 * 64 + 32 + kc0];
    pa[1].q[1] = *(const uint4*)&pw[l16 * 64 + 32 + kc1];

#pragma unroll
    for (int dt = 0; dt < 4; ++dt) {  // O += P V  (contract over 64 keys)
      Frag bv0, bv1;
      int dr = dt * 16 + l16;
      bv0.q[0] = *(const uint4*)&vbuf[dr * 64 + kc0];
      bv0.q[1] = *(const uint4*)&vbuf[dr * 64 + kc1];
      bv1.q[0] = *(const uint4*)&vbuf[dr * 64 + 32 + kc0];
      bv1.q[1] = *(const uint4*)&vbuf[dr * 64 + 32 + kc1];
      o_acc[dt] = __builtin_amdgcn_wmma_f32_16x16x32_bf16(
          false, pa[0].v, false, bv0.v, (short)0, o_acc[dt], false, false);
      o_acc[dt] = __builtin_amdgcn_wmma_f32_16x16x32_bf16(
          false, pa[1].v, false, bv1.v, (short)0, o_acc[dt], false, false);
    }
  }

#pragma unroll
  for (int dt = 0; dt < 4; ++dt) {
#pragma unroll
    for (int r = 0; r < 8; ++r) {
      int srow = qr0 + r + 8 * half;
      int col = h * DK + dt * 16 + l16;
      Y[(size_t)(bb * S_ + srow) * DM + col] = f2bf(o_acc[dt][r] / Lrow[r]);
    }
  }
}

extern "C" void kernel_launch(void* const* d_in, const int* in_sizes, int n_in,
                              void* d_out, int out_size, void* d_ws, size_t ws_size,
                              hipStream_t stream) {
  const float* query  = (const float*)d_in[0];
  const float* keys   = (const float*)d_in[1];
  const float* values = (const float*)d_in[2];
  // d_in[3] = mask (causal, implemented analytically)
  const float* Wq = (const float*)d_in[4];
  const float* bq = (const float*)d_in[5];
  const float* Wk = (const float*)d_in[6];
  const float* bk = (const float*)d_in[7];
  const float* Wv = (const float*)d_in[8];
  const float* bv = (const float*)d_in[9];
  const float* Wo = (const float*)d_in[10];
  const float* bo = (const float*)d_in[11];
  float* out = (float*)d_out;

  unsigned char* ws = (unsigned char*)d_ws;
  const size_t WSZ = (size_t)1024 * 1024 * 2;  // bf16 weight: 2 MiB
  const size_t XSZ = (size_t)4096 * 1024 * 2;  // bf16 activation: 8 MiB
  unsigned short* wWq = (unsigned short*)(ws);
  unsigned short* wWk = (unsigned short*)(ws + WSZ);
  unsigned short* wWv = (unsigned short*)(ws + 2 * WSZ);
  unsigned short* wWo = (unsigned short*)(ws + 3 * WSZ);
  unsigned short* xq  = (unsigned short*)(ws + 4 * WSZ);
  unsigned short* xk  = (unsigned short*)(ws + 4 * WSZ + XSZ);
  unsigned short* xv  = (unsigned short*)(ws + 4 * WSZ + 2 * XSZ);
  unsigned short* Qb  = (unsigned short*)(ws + 4 * WSZ + 3 * XSZ);
  unsigned short* Kb  = (unsigned short*)(ws + 4 * WSZ + 4 * XSZ);
  unsigned short* Vtb = (unsigned short*)(ws + 4 * WSZ + 5 * XSZ);
  unsigned short* Yb  = (unsigned short*)(ws + 4 * WSZ + 6 * XSZ);

  const int nW = 1024 * 1024, nX = 4096 * 1024;
  cvt_bf16_kernel<<<512, 256, 0, stream>>>(query,  xq,  nX);
  cvt_bf16_kernel<<<512, 256, 0, stream>>>(keys,   xk,  nX);
  cvt_bf16_kernel<<<512, 256, 0, stream>>>(values, xv,  nX);
  cvt_bf16_kernel<<<256, 256, 0, stream>>>(Wq, wWq, nW);
  cvt_bf16_kernel<<<256, 256, 0, stream>>>(Wk, wWk, nW);
  cvt_bf16_kernel<<<256, 256, 0, stream>>>(Wv, wWv, nW);
  cvt_bf16_kernel<<<256, 256, 0, stream>>>(Wo, wWo, nW);

  dim3 ggrid(8, 32);
  gemm_qkv_kernel<<<ggrid, 256, 0, stream>>>(xq, wWq, bq, nullptr, Qb, 1);
  gemm_qkv_kernel<<<ggrid, 256, 0, stream>>>(xk, wWk, bk, nullptr, Kb, 1);
  gemm_qkv_kernel<<<ggrid, 256, 0, stream>>>(xv, wWv, bv, nullptr, Vtb, 2);

  flash_attn_kernel<<<dim3(16, 64), 128, 0, stream>>>(Qb, Kb, Vtb, Yb);

  gemm_qkv_kernel<<<ggrid, 256, 0, stream>>>(Yb, wWo, bo, out, nullptr, 0);
}
